// Block_21028159881813
// MI455X (gfx1250) — matrix operand order repaired and checked by reading, not compiled, not run on hardware
//
#include <hip/hip_runtime.h>

// ---------------------------------------------------------------------------
// MI455X (gfx1250) efficient-attention transformer block.
// All GEMMs: v_wmma_f32_16x16x32_bf16, fp32 accumulate, bf16 operands.
// Block = 256 threads = 8 waves (wave32); each wave owns a 16 x (NTILES*16)
// output tile; waves split the M dimension only, so every tile-guard is a
// scalar (blockIdx-derived) condition -> no EXEC masking around WMMA.
// Inner loop: load ALL B fragments into distinct registers, then issue the
// WMMA group, so s_wait_loadcnt becomes partial and load latency overlaps
// with matrix ops. Row pointers are strength-reduced (no per-iter v_mul_u64).
// ---------------------------------------------------------------------------

typedef __bf16 bf16_t;
typedef bf16_t v16bf __attribute__((ext_vector_type(16)));
typedef bf16_t v8bf  __attribute__((ext_vector_type(8)));
typedef float  v8f   __attribute__((ext_vector_type(8)));

// Problem constants (match reference).
#define Nb 16
#define Lx 3136
#define Dh 768
#define Hh 8
#define Kc 768
#define Vc 384
#define Mm 3072
#define HK 96
#define HV 48

__device__ __forceinline__ bf16_t f2bf(float f) {
  // round-to-nearest-even fp32 -> bf16 via integer ops
  unsigned u = __builtin_bit_cast(unsigned, f);
  unsigned r = (u + 0x7FFFu + ((u >> 16) & 1u)) >> 16;
  return __builtin_bit_cast(bf16_t, (unsigned short)r);
}

__device__ __forceinline__ bf16_t bf_zero() {
  return __builtin_bit_cast(bf16_t, (unsigned short)0);
}

__device__ __forceinline__ v16bf combine16(v8bf lo, v8bf hi) {
  v16bf f;
#pragma unroll
  for (int i = 0; i < 8; ++i) { f[i] = lo[i]; f[i + 8] = hi[i]; }
  return f;
}

// Generic batched WMMA GEMM:  C[z] = act(A[z] * B[z] + bias) (+ residual)
// NTILES: 16-col subtiles per wave (grid.x * NTILES*16 == Ncols exactly).
// MGUARD: 1 if Mrows is not a multiple of 128 (row guards needed).
// OUT_MODE: 0 = f32, 1 = bf16, 2 = bf16 transposed store, 3 = f32 + residual.
// BIAS_MODE: 0 = none, 1 = per-row, 2 = per-col.  ACT: 0 = none, 1 = exact gelu.
// Batch offset for pointer P is (z/HB)*POuter + (z%HB)*PInner (element units).
//
// Fragment layouts (CDNA5 ISA):
//  A (16x32 bf16): lane&15 = row; lane>>4 selects K halves {0..7,16..23} vs
//                  {8..15,24..31} -> two contiguous 16B loads per lane.
//  B (32x16 bf16): lane&15 = col; lane>>4 picks the same K halves as A (ldb-strided loads).
//  C/D (16x16 f32): element r <-> row = r + 8*(lane>=16), col = lane&15.
template <int NTILES, int MGUARD, int OUT_MODE, int BIAS_MODE, int ACT>
__global__ void __launch_bounds__(256)
wmma_gemm(const bf16_t* __restrict__ A, const bf16_t* __restrict__ B,
          const float* __restrict__ bias, const float* __restrict__ Res,
          float* __restrict__ Cf, bf16_t* __restrict__ Cb,
          int Mrows, int Kdim, int lda, int ldb, int ldc, int HB,
          long aOuter, long aInner, long bOuter, long bInner,
          long cOuter, long cInner, long rOuter, long rInner) {
  const int z = blockIdx.z;
  const int zo = z / HB, zi = z % HB;
  A += zo * aOuter + zi * aInner;
  B += zo * bOuter + zi * bInner;
  const long cOff = zo * cOuter + zi * cInner;
  if (OUT_MODE == 0 || OUT_MODE == 3) Cf += cOff; else Cb += cOff;
  if (OUT_MODE == 3) Res += zo * rOuter + zi * rInner;

  const int wave = threadIdx.x >> 5;
  const int lane = threadIdx.x & 31;
  const int mBase = blockIdx.y * 128 + wave * 16;   // waves split M only
  const int nBase = blockIdx.x * (NTILES * 16);     // scalar (SGPR) N base

  // Per-lane row pointers, strength-reduced across the K loop.
  const int aRowIdx = mBase + (lane & 15);
  const bool aValid = !MGUARD || (aRowIdx < Mrows);
  const bf16_t* aRow = A + (long)aRowIdx * lda + ((lane >> 4) << 3);
  const bf16_t* bRow = B + (long)((lane >> 4) << 3) * ldb + nBase + (lane & 15);
  const long bStep = (long)32 * ldb;

  v8f acc[NTILES];
#pragma unroll
  for (int j = 0; j < NTILES; ++j)
#pragma unroll
    for (int r = 0; r < 8; ++r) acc[j][r] = 0.0f;

  for (int kk = 0; kk < Kdim; kk += 32, aRow += 32, bRow += bStep) {
    if (kk + 32 < Kdim)  // speculative near prefetch of next B K-slab
      __builtin_prefetch(bRow + bStep, 0, 3);

    // Issue ALL fragment loads first (distinct registers), then the WMMA
    // group: loads complete in order, so each WMMA only waits for its own.
    v16bf b[NTILES];
#pragma unroll
    for (int j = 0; j < NTILES; ++j)
#pragma unroll
      for (int i = 0; i < 8; ++i) {
        b[j][i] = bRow[(long)i * ldb + j * 16];
        b[j][i + 8] = bRow[(long)(i + 16) * ldb + j * 16];
      }
    v16bf a;
    if (aValid) {
      a = combine16(*(const v8bf*)(aRow), *(const v8bf*)(aRow + 16));
    } else {
#pragma unroll
      for (int i = 0; i < 16; ++i) a[i] = bf_zero();
    }
#pragma unroll
    for (int j = 0; j < NTILES; ++j)
      acc[j] = __builtin_amdgcn_wmma_f32_16x16x32_bf16(
          false, a, false, b[j], (short)0, acc[j], false, false);
  }

  // C/D layout: element r <-> row = r + 8*(lane>=16), col = lane&15
  const int rBase = mBase + ((lane >> 4) << 3);
  const int col0 = lane & 15;
#pragma unroll
  for (int j = 0; j < NTILES; ++j) {
    const int col = nBase + j * 16 + col0;
#pragma unroll
    for (int r = 0; r < 8; ++r) {
      const int row = rBase + r;
      if (MGUARD && row >= Mrows) continue;
      float v = acc[j][r];
      if (BIAS_MODE == 1) v += bias[row];
      if (BIAS_MODE == 2) v += bias[col];
      if (ACT == 1) v = 0.5f * v * (1.0f + erff(v * 0.70710678118654752f));
      if (OUT_MODE == 0) Cf[(long)row * ldc + col] = v;
      else if (OUT_MODE == 1) Cb[(long)row * ldc + col] = f2bf(v);
      else if (OUT_MODE == 2) Cb[(long)col * ldc + row] = f2bf(v);
      else Cf[(long)row * ldc + col] = v + Res[(long)row * ldc + col];
    }
  }
}

// LayerNorm over last dim (row of length Dm), bf16 output. One block per row.
__global__ void __launch_bounds__(256)
layernorm_bf16(const float* __restrict__ x, const float* __restrict__ g,
               const float* __restrict__ b, bf16_t* __restrict__ out, int Dm) {
  __shared__ float redS[8], redQ[8], stat[2];
  const long base = (long)blockIdx.x * Dm;
  const int lane = threadIdx.x & 31, wave = threadIdx.x >> 5;
  float s = 0.f, q = 0.f;
  for (int i = threadIdx.x; i < Dm; i += 256) { float v = x[base + i]; s += v; q += v * v; }
  for (int o = 16; o; o >>= 1) { s += __shfl_xor(s, o, 32); q += __shfl_xor(q, o, 32); }
  if (lane == 0) { redS[wave] = s; redQ[wave] = q; }
  __syncthreads();
  if (threadIdx.x == 0) {
    float ts = 0.f, tq = 0.f;
    for (int w = 0; w < 8; ++w) { ts += redS[w]; tq += redQ[w]; }
    float m = ts / Dm;
    stat[0] = m;
    stat[1] = rsqrtf(tq / Dm - m * m + 1e-6f);
  }
  __syncthreads();
  const float m = stat[0], inv = stat[1];
  for (int i = threadIdx.x; i < Dm; i += 256)
    out[base + i] = f2bf((x[base + i] - m) * inv * g[i] + b[i]);
}

// Softmax over a row of length Lm (keys: softmax over spatial). Block per row.
__global__ void __launch_bounds__(256)
softmax_rows_bf16(const float* __restrict__ in, bf16_t* __restrict__ out, int Lm) {
  __shared__ float red[8], stat[2];
  const long base = (long)blockIdx.x * Lm;
  const int lane = threadIdx.x & 31, wave = threadIdx.x >> 5;
  float mx = -3.402823466e38f;
  for (int i = threadIdx.x; i < Lm; i += 256) mx = fmaxf(mx, in[base + i]);
  for (int o = 16; o; o >>= 1) mx = fmaxf(mx, __shfl_xor(mx, o, 32));
  if (lane == 0) red[wave] = mx;
  __syncthreads();
  if (threadIdx.x == 0) {
    float m = red[0];
    for (int w = 1; w < 8; ++w) m = fmaxf(m, red[w]);
    stat[0] = m;
  }
  __syncthreads();
  mx = stat[0];
  float s = 0.f;
  for (int i = threadIdx.x; i < Lm; i += 256) s += __expf(in[base + i] - mx);
  for (int o = 16; o; o >>= 1) s += __shfl_xor(s, o, 32);
  if (lane == 0) red[wave] = s;
  __syncthreads();
  if (threadIdx.x == 0) {
    float t = 0.f;
    for (int w = 0; w < 8; ++w) t += red[w];
    stat[1] = 1.0f / t;
  }
  __syncthreads();
  const float inv = stat[1];
  for (int i = threadIdx.x; i < Lm; i += 256)
    out[base + i] = f2bf(__expf(in[base + i] - mx) * inv);
}

// Softmax over the 96 head-channels (stride Lm) for each (n, h, l).
__global__ void __launch_bounds__(256)
softmax_heads_bf16(const float* __restrict__ q, bf16_t* __restrict__ out,
                   int total, int Lm, int hk) {
  const int t = blockIdx.x * 256 + threadIdx.x;
  if (t >= total) return;
  const int l = t % Lm;
  const long nh = t / Lm;
  const long base = nh * (long)hk * Lm + l;
  float mx = -3.402823466e38f;
  for (int c = 0; c < hk; ++c) mx = fmaxf(mx, q[base + (long)c * Lm]);
  float s = 0.f;
  for (int c = 0; c < hk; ++c) s += __expf(q[base + (long)c * Lm] - mx);
  const float inv = 1.0f / s;
  for (int c = 0; c < hk; ++c)
    out[base + (long)c * Lm] = f2bf(__expf(q[base + (long)c * Lm] - mx) * inv);
}

__global__ void __launch_bounds__(256)
cvt_f32_bf16(const float* __restrict__ in, bf16_t* __restrict__ out, long n) {
  long i = (long)blockIdx.x * 256 + threadIdx.x;
  if (i < n) out[i] = f2bf(in[i]);
}

__global__ void __launch_bounds__(256)
transpose_f32_bf16(const float* __restrict__ in, bf16_t* __restrict__ out, int R, int C) {
  long i = (long)blockIdx.x * 256 + threadIdx.x;
  if (i < (long)R * C) {
    int r = (int)(i / C), c = (int)(i % C);
    out[(long)c * R + r] = f2bf(in[i]);
  }
}

static inline long cdiv(long a, long b) { return (a + b - 1) / b; }

extern "C" void kernel_launch(void* const* d_in, const int* in_sizes, int n_in,
                              void* d_out, int out_size, void* d_ws, size_t ws_size,
                              hipStream_t stream) {
  (void)in_sizes; (void)n_in; (void)out_size; (void)ws_size;
  const float* x     = (const float*)d_in[0];
  const float* ln1_g = (const float*)d_in[1];
  const float* ln1_b = (const float*)d_in[2];
  const float* Wk    = (const float*)d_in[3];
  const float* bk    = (const float*)d_in[4];
  const float* Wq    = (const float*)d_in[5];
  const float* bq    = (const float*)d_in[6];
  const float* Wv    = (const float*)d_in[7];
  const float* bv    = (const float*)d_in[8];
  const float* Wr    = (const float*)d_in[9];
  const float* br    = (const float*)d_in[10];
  const float* ln2_g = (const float*)d_in[11];
  const float* ln2_b = (const float*)d_in[12];
  const float* W1    = (const float*)d_in[13];
  const float* b1    = (const float*)d_in[14];
  const float* W2    = (const float*)d_in[15];
  const float* b2    = (const float*)d_in[16];
  float* out = (float*)d_out;

  const long NLD = (long)Nb * Lx * Dh;
  size_t off = 0;
  auto carve = [&](size_t bytes) {
    void* p = (char*)d_ws + off;
    off += (bytes + 255) & ~(size_t)255;
    return p;
  };
  bf16_t* yb    = (bf16_t*)carve(NLD * 2);                       // LN1 out, raw [D,L] view
  bf16_t* WkB   = (bf16_t*)carve((size_t)Kc * Dh * 2);
  bf16_t* WqB   = (bf16_t*)carve((size_t)Kc * Dh * 2);
  bf16_t* WvB   = (bf16_t*)carve((size_t)Vc * Dh * 2);
  bf16_t* WrB   = (bf16_t*)carve((size_t)Dh * Vc * 2);
  bf16_t* W1TB  = (bf16_t*)carve((size_t)Dh * Mm * 2);           // W1^T [D,M]
  bf16_t* W2TB  = (bf16_t*)carve((size_t)Mm * Dh * 2);           // W2^T [M,D]
  float*  keysF = (float*)carve((size_t)Nb * Kc * Lx * 4);
  float*  quesF = (float*)carve((size_t)Nb * Kc * Lx * 4);
  bf16_t* kh    = (bf16_t*)carve((size_t)Nb * Kc * Lx * 2);
  bf16_t* qh    = (bf16_t*)carve((size_t)Nb * Kc * Lx * 2);
  bf16_t* vT    = (bf16_t*)carve((size_t)Nb * Lx * Vc * 2);      // values^T [L,V] per n
  bf16_t* ctxT  = (bf16_t*)carve((size_t)Nb * Hh * HV * HK * 2); // context^T [48,96] per (n,h)
  bf16_t* agg   = (bf16_t*)carve((size_t)Nb * Vc * Lx * 2);      // att [V,L] per n
  float*  x1    = (float*)carve(NLD * 4);                        // after first residual (flat)
  bf16_t* y2    = (bf16_t*)carve(NLD * 2);                       // LN2 out, [NL,D]
  bf16_t* hid   = (bf16_t*)carve((size_t)Nb * Lx * Mm * 2);      // MLP hidden

  const dim3 blk(256);

  // ---- weight conversion / transposition (cheap, once per call) ----
  cvt_f32_bf16<<<dim3(cdiv((long)Kc * Dh, 256)), blk, 0, stream>>>(Wk, WkB, (long)Kc * Dh);
  cvt_f32_bf16<<<dim3(cdiv((long)Kc * Dh, 256)), blk, 0, stream>>>(Wq, WqB, (long)Kc * Dh);
  cvt_f32_bf16<<<dim3(cdiv((long)Vc * Dh, 256)), blk, 0, stream>>>(Wv, WvB, (long)Vc * Dh);
  cvt_f32_bf16<<<dim3(cdiv((long)Dh * Vc, 256)), blk, 0, stream>>>(Wr, WrB, (long)Dh * Vc);
  transpose_f32_bf16<<<dim3(cdiv((long)Mm * Dh, 256)), blk, 0, stream>>>(W1, W1TB, Mm, Dh);
  transpose_f32_bf16<<<dim3(cdiv((long)Dh * Mm, 256)), blk, 0, stream>>>(W2, W2TB, Dh, Mm);

  // ---- LN1: [N*L] rows of D ----
  layernorm_bf16<<<dim3(Nb * Lx), blk, 0, stream>>>(x, ln1_g, ln1_b, yb, Dh);

  // ---- projections: C[o,l] = W[o,:] . yb_n[:,l]  (yb raw-viewed as [D,L]) ----
  // grids: x = N/64, y = M/128, z = batch
  wmma_gemm<4, 0, 0, 1, 0><<<dim3(49, 6, Nb), blk, 0, stream>>>(
      WkB, yb, bk, nullptr, keysF, nullptr, Kc, Dh, Dh, Lx, Lx,
      1, 0, 0, (long)Dh * Lx, 0, (long)Kc * Lx, 0, 0, 0);
  wmma_gemm<4, 0, 0, 1, 0><<<dim3(49, 6, Nb), blk, 0, stream>>>(
      WqB, yb, bq, nullptr, quesF, nullptr, Kc, Dh, Dh, Lx, Lx,
      1, 0, 0, (long)Dh * Lx, 0, (long)Kc * Lx, 0, 0, 0);
  // values stored transposed as vT[l, v] so the context GEMM's B is contiguous
  wmma_gemm<4, 0, 2, 1, 0><<<dim3(49, 3, Nb), blk, 0, stream>>>(
      WvB, yb, bv, nullptr, nullptr, vT, Vc, Dh, Dh, Lx, Vc,
      1, 0, 0, (long)Dh * Lx, 0, (long)Lx * Vc, 0, 0, 0);

  // ---- softmaxes ----
  softmax_rows_bf16<<<dim3(Nb * Kc), blk, 0, stream>>>(keysF, kh, Lx);        // over L
  softmax_heads_bf16<<<dim3((Nb * Hh * Lx) / 256), blk, 0, stream>>>(         // over 96 ch
      quesF, qh, Nb * Hh * Lx, Lx, HK);

  // ---- context[k,v] = kh[k,:] . vh[v,:] per (n,h); stored transposed [48,96] ----
  // M=96 (guarded within one 128 block), N=48 -> NTILES=3
  wmma_gemm<3, 1, 2, 0, 0><<<dim3(1, 1, Nb * Hh), blk, 0, stream>>>(
      kh, vT, nullptr, nullptr, nullptr, ctxT, HK, Lx, Lx, Vc, HK,
      Hh, (long)Kc * Lx, (long)HK * Lx, (long)Lx * Vc, (long)HV,
      (long)Hh * HV * HK, (long)HV * HK, 0, 0);

  // ---- att[v,l] = ctxT[v,:] . qh[:,l] per (n,h) -> agg[V,L] per n ----
  wmma_gemm<4, 1, 1, 0, 0><<<dim3(49, 1, Nb * Hh), blk, 0, stream>>>(
      ctxT, qh, nullptr, nullptr, nullptr, agg, HV, HK, HK, Lx, Lx,
      Hh, (long)Hh * HV * HK, (long)HV * HK, (long)Kc * Lx, (long)HK * Lx,
      (long)Vc * Lx, (long)HV * Lx, 0, 0);

  // ---- Wr projection + raw-flat residual: x1 = Wr.agg + br + x (flat add) ----
  wmma_gemm<4, 0, 3, 1, 0><<<dim3(49, 6, Nb), blk, 0, stream>>>(
      WrB, agg, br, x, x1, nullptr, Dh, Vc, Vc, Lx, Lx,
      1, 0, 0, (long)Vc * Lx, 0, (long)Lx * Dh, 0, (long)Lx * Dh, 0);

  // ---- LN2 over x1 rows [NL, D] ----
  layernorm_bf16<<<dim3(Nb * Lx), blk, 0, stream>>>(x1, ln2_g, ln2_b, y2, Dh);

  // ---- MLP: hid = gelu(y2 @ W1^T + b1) ; out = hid @ W2^T + b2 + x1 ----
  wmma_gemm<4, 0, 1, 2, 1><<<dim3(48, 392, 1), blk, 0, stream>>>(
      y2, W1TB, b1, nullptr, nullptr, hid, Nb * Lx, Dh, Dh, Mm, Mm,
      1, 0, 0, 0, 0, 0, 0, 0, 0);
  wmma_gemm<4, 0, 3, 2, 0><<<dim3(12, 392, 1), blk, 0, stream>>>(
      hid, W2TB, b2, x1, out, nullptr, Nb * Lx, Mm, Mm, Dh, Dh,
      1, 0, 0, 0, 0, 0, 0, 0, 0);
}
